// BiMambaTextEncoder_57612691309403
// MI455X (gfx1250) — matrix operand order repaired and checked by reading, not compiled, not run on hardware
//
#include <hip/hip_runtime.h>
#include <hip/hip_bf16.h>

// ---------------------------------------------------------------------------
// BiMamba text encoder for MI455X (gfx1250): bf16 WMMA GEMMs (async-LDS
// double-buffered, 2x2 register blocking) + VALU SSM scan
// ---------------------------------------------------------------------------

typedef __attribute__((ext_vector_type(16))) __bf16 v16bf;
typedef __attribute__((ext_vector_type(8)))  float  v8f;

#define B_     4
#define L_     1024
#define C_     512
#define KT_    5
#define DEPTH_ 3
#define DI_    1024
#define DS_    16
#define DC_    4
#define DTR_   32
#define NBL    (B_ * L_)

// GEMM tiling
#define BM_    64
#define BN_    128
#define BK_    32
#define AELEMS (BM_ * BK_)            // 2048 bf16
#define BELEMS (BN_ * BK_)            // 4096 bf16
#define BUFEL  (AELEMS + BELEMS)      // 6144 bf16 per buffer
#define GEMM_LDS_BYTES (2 * BUFEL * 2)

// ---------------------- CDNA5 async LDS fill (ASYNCcnt) --------------------
__device__ inline void async_ld_b128(unsigned lds_byte_off, const void* gptr) {
    unsigned long long ga = (unsigned long long)(uintptr_t)gptr;
    asm volatile("global_load_async_to_lds_b128 %0, %1, off"
                 :: "v"(lds_byte_off), "v"(ga) : "memory");
}
__device__ inline void wait_async0() {
    asm volatile("s_wait_asynccnt 0x0" ::: "memory");
}

// ------------------------- WMMA fragment loaders ---------------------------
// A fragment: 16x32 bf16 tile (row-major). Per-lane data is two contiguous
// 16B runs -> compiler emits 2x ds_load_b128 (verified in disasm).
__device__ inline v16bf frag_a(const __bf16* __restrict__ tile, int lane) {
    const int m  = lane & 15;
    const int kb = (lane & 16) ? 8 : 0;
    const __bf16* row = tile + m * 32;
    v16bf f;
#pragma unroll
    for (int j = 0; j < 8; ++j) {
        const int k = (j < 4) ? (kb + 2 * j) : (kb + 8 + 2 * j); // j>=4 -> +16
        f[2 * j]     = row[k];
        f[2 * j + 1] = row[k + 1];
    }
    return f;
}

// B fragment: 32x16 tile stored as Bt [n][k]; two contiguous 16B runs per lane.
__device__ inline v16bf frag_b(const __bf16* __restrict__ tile, int lane) {
    const int n  = lane & 15;
    const int kb = (lane & 16) ? 8 : 0;
    const __bf16* row = tile + n * 32;
    v16bf f;
#pragma unroll
    for (int j = 0; j < 8; ++j) {
        const int k = (j < 4) ? (kb + 2 * j) : (kb + 8 + 2 * j);
        f[2 * j]     = row[k];
        f[2 * j + 1] = row[k + 1];
    }
    return f;
}

// ------------------------- generic bf16 GEMM -------------------------------
// out[n][m] = sum_k A[m][k] * Bt[n][k]  (+bias[m], optional softplus)
// grid = (M/64, N/128), block = 256 threads (8 waves), 2x2 WMMA per wave,
// GLOBAL_LOAD_ASYNC_TO_LDS_B128 double buffering.
__global__ __launch_bounds__(256)
void gemm_bt_wmma(const __bf16* __restrict__ A, int lda,
                  const __bf16* __restrict__ Bt, int ldb,
                  const float* __restrict__ bias,
                  float* __restrict__ outF, __bf16* __restrict__ outBF,
                  int M, int N, int K, int act) {
    extern __shared__ __align__(16) __bf16 sm[];   // only dynamic LDS -> offset 0
    const int tid  = threadIdx.x;
    const int lane = tid & 31;
    const int wave = tid >> 5;
    const int m0 = blockIdx.x * BM_;
    const int n0 = blockIdx.y * BN_;
    const int mq = (wave & 1) * 32;
    const int nq = (wave >> 1) * 32;
    const int row = tid >> 2;            // 0..63
    const int seg = (tid & 3) * 8;       // bf16 elements, 16B granules

    v8f acc00 = {}, acc01 = {}, acc10 = {}, acc11 = {};

    const __bf16* gA  = A  + (size_t)(m0 + row) * lda + seg;
    const __bf16* gB0 = Bt + (size_t)(n0 + row) * ldb + seg;
    const __bf16* gB1 = Bt + (size_t)(n0 + row + 64) * ldb + seg;
    const unsigned aOff  = (unsigned)((row * 32 + seg) * 2);
    const unsigned bOff0 = (unsigned)(AELEMS * 2 + (row * 32 + seg) * 2);
    const unsigned bOff1 = (unsigned)(AELEMS * 2 + ((row + 64) * 32 + seg) * 2);

    // prologue: fill buffer 0 with K-chunk 0
    async_ld_b128(aOff,  gA);
    async_ld_b128(bOff0, gB0);
    async_ld_b128(bOff1, gB1);
    wait_async0();
    __syncthreads();

    int buf = 0;
    for (int k0 = 0; k0 < K; k0 += BK_) {
        if (k0 + BK_ < K) {
            const unsigned bb = (unsigned)((buf ^ 1) * BUFEL * 2);
            async_ld_b128(bb + aOff,  gA  + k0 + BK_);
            async_ld_b128(bb + bOff0, gB0 + k0 + BK_);
            async_ld_b128(bb + bOff1, gB1 + k0 + BK_);
        }
        const __bf16* As = sm + buf * BUFEL;
        const __bf16* Bs = As + AELEMS;
        v16bf fa0 = frag_a(As + (mq +  0) * 32, lane);
        v16bf fa1 = frag_a(As + (mq + 16) * 32, lane);
        v16bf fb0 = frag_b(Bs + (nq +  0) * 32, lane);
        v16bf fb1 = frag_b(Bs + (nq + 16) * 32, lane);
        acc00 = __builtin_amdgcn_wmma_f32_16x16x32_bf16(false, fa0, false, fb0, (short)0, acc00, false, false);
        acc01 = __builtin_amdgcn_wmma_f32_16x16x32_bf16(false, fa0, false, fb1, (short)0, acc01, false, false);
        acc10 = __builtin_amdgcn_wmma_f32_16x16x32_bf16(false, fa1, false, fb0, (short)0, acc10, false, false);
        acc11 = __builtin_amdgcn_wmma_f32_16x16x32_bf16(false, fa1, false, fb1, (short)0, acc11, false, false);
        wait_async0();
        __syncthreads();
        buf ^= 1;
    }

    const int lane15 = lane & 15;
    const int mb = (lane & 16) ? 8 : 0;
#pragma unroll
    for (int ni = 0; ni < 2; ++ni) {
        const int n = n0 + nq + ni * 16 + lane15;
#pragma unroll
        for (int mi = 0; mi < 2; ++mi) {
            const v8f a = ni == 0 ? (mi == 0 ? acc00 : acc10)
                                  : (mi == 0 ? acc01 : acc11);
#pragma unroll
            for (int r = 0; r < 8; ++r) {
                const int m = m0 + mq + mi * 16 + mb + r;
                float v = a[r];
                if (bias) v += bias[m];
                if (act == 1) v = (v > 20.f) ? v : log1pf(__expf(v));  // softplus
                const size_t o = (size_t)n * M + m;
                if (outF)  outF[o]  = v;
                if (outBF) outBF[o] = (__bf16)v;
            }
        }
    }
}

// ------------------------- implicit-GEMM conv1d ----------------------------
// out[b][l][cout] = bias[cout] + sum_{tap,cin} W[tap][cout][cin]*X[b][l+tap-2][cin]
// Same async 8-wave 2x2 design as the GEMM; K sequence = 80 chunks of
// (tap 0..4) x (cin chunks of 32). Boundary rows: async-load a clamped row,
// then overwrite with zeros in LDS after s_wait_asynccnt.
__global__ __launch_bounds__(256)
void conv1d_wmma(const __bf16* __restrict__ W,   // [KT][C][C] bf16
                 const __bf16* __restrict__ X,   // [B][L][C]  bf16
                 const float* __restrict__ bias, // [C]
                 float* __restrict__ outF) {     // [B][L][C]  f32
    extern __shared__ __align__(16) __bf16 sm[];
    const int tid  = threadIdx.x;
    const int lane = tid & 31;
    const int wave = tid >> 5;
    const int m0 = blockIdx.x * BM_;     // cout tile
    const int l0 = blockIdx.y * BN_;     // L tile
    const int b  = blockIdx.z;
    const int mq = (wave & 1) * 32;
    const int nq = (wave >> 1) * 32;
    const int row = tid >> 2;            // 0..63
    const int seg = (tid & 3) * 8;
    const unsigned aOff  = (unsigned)((row * 32 + seg) * 2);
    const unsigned bOff0 = (unsigned)(AELEMS * 2 + (row * 32 + seg) * 2);
    const unsigned bOff1 = (unsigned)(AELEMS * 2 + ((row + 64) * 32 + seg) * 2);
    const int NCHUNK = KT_ * (C_ / BK_);   // 80

    v8f acc00 = {}, acc01 = {}, acc10 = {}, acc11 = {};

    auto stage = [&](int q, int bufi) {
        const int tap = q >> 4;
        const int c0  = (q & 15) * BK_;
        const unsigned bb = (unsigned)(bufi * BUFEL * 2);
        async_ld_b128(bb + aOff,
                      W + (size_t)tap * C_ * C_ + (size_t)(m0 + row) * C_ + c0 + seg);
        int lp0 = l0 + row + tap - (KT_ - 1) / 2;
        int lp1 = lp0 + 64;
        const int cl0 = lp0 < 0 ? 0 : (lp0 >= L_ ? L_ - 1 : lp0);
        const int cl1 = lp1 < 0 ? 0 : (lp1 >= L_ ? L_ - 1 : lp1);
        async_ld_b128(bb + bOff0, X + ((size_t)b * L_ + cl0) * C_ + c0 + seg);
        async_ld_b128(bb + bOff1, X + ((size_t)b * L_ + cl1) * C_ + c0 + seg);
    };
    auto fix_zero = [&](int q, int bufi) {   // call after wait_async0()
        const int tap = q >> 4;
        const int lp0 = l0 + row + tap - (KT_ - 1) / 2;
        const int lp1 = lp0 + 64;
        const uint4 z = {0u, 0u, 0u, 0u};
        __bf16* Bb = sm + bufi * BUFEL + AELEMS;
        if (lp0 < 0 || lp0 >= L_) *(uint4*)(Bb + row * 32 + seg) = z;
        if (lp1 < 0 || lp1 >= L_) *(uint4*)(Bb + (row + 64) * 32 + seg) = z;
    };

    stage(0, 0);
    wait_async0();
    fix_zero(0, 0);
    __syncthreads();

    int buf = 0;
    for (int q = 0; q < NCHUNK; ++q) {
        if (q + 1 < NCHUNK) stage(q + 1, buf ^ 1);
        const __bf16* As = sm + buf * BUFEL;
        const __bf16* Bs = As + AELEMS;
        v16bf fa0 = frag_a(As + (mq +  0) * 32, lane);
        v16bf fa1 = frag_a(As + (mq + 16) * 32, lane);
        v16bf fb0 = frag_b(Bs + (nq +  0) * 32, lane);
        v16bf fb1 = frag_b(Bs + (nq + 16) * 32, lane);
        acc00 = __builtin_amdgcn_wmma_f32_16x16x32_bf16(false, fa0, false, fb0, (short)0, acc00, false, false);
        acc01 = __builtin_amdgcn_wmma_f32_16x16x32_bf16(false, fa0, false, fb1, (short)0, acc01, false, false);
        acc10 = __builtin_amdgcn_wmma_f32_16x16x32_bf16(false, fa1, false, fb0, (short)0, acc10, false, false);
        acc11 = __builtin_amdgcn_wmma_f32_16x16x32_bf16(false, fa1, false, fb1, (short)0, acc11, false, false);
        wait_async0();
        if (q + 1 < NCHUNK) fix_zero(q + 1, buf ^ 1);
        __syncthreads();
        buf ^= 1;
    }

    const int lane15 = lane & 15;
    const int mb = (lane & 16) ? 8 : 0;
#pragma unroll
    for (int ni = 0; ni < 2; ++ni) {
        const size_t bl = (size_t)b * L_ + l0 + nq + ni * 16 + lane15;
#pragma unroll
        for (int mi = 0; mi < 2; ++mi) {
            const v8f a = ni == 0 ? (mi == 0 ? acc00 : acc10)
                                  : (mi == 0 ? acc01 : acc11);
#pragma unroll
            for (int r = 0; r < 8; ++r) {
                const int m = m0 + mq + mi * 16 + mb + r;
                outF[bl * C_ + m] = a[r] + bias[m];
            }
        }
    }
}

// ------------------------- elementwise / scan kernels ----------------------
__global__ void embed_kernel(const int* __restrict__ x, const float* __restrict__ emb,
                             __bf16* __restrict__ h) {
    const int idx = blockIdx.x * blockDim.x + threadIdx.x;  // B*L*C
    const int c  = idx & (C_ - 1);
    const int bl = idx >> 9;
    h[idx] = (__bf16)emb[(size_t)x[bl] * C_ + c];
}

__global__ void f32_to_bf16_kernel(const float* __restrict__ s, __bf16* __restrict__ d, int n) {
    const int idx = blockIdx.x * blockDim.x + threadIdx.x;
    if (idx < n) d[idx] = (__bf16)s[idx];
}

// conv_w [depth][cout][cin][k] f32  ->  [depth][k][cout][cin] bf16
__global__ void pack_conv_w_kernel(const float* __restrict__ w, __bf16* __restrict__ o) {
    const int idx = blockIdx.x * blockDim.x + threadIdx.x;  // DEPTH*C*C*KT
    if (idx >= DEPTH_ * C_ * C_ * KT_) return;
    const int k    = idx % KT_;
    const int cin  = (idx / KT_) % C_;
    const int cout = (idx / (KT_ * C_)) % C_;
    const int d    = idx / (KT_ * C_ * C_);
    o[(((size_t)d * KT_ + k) * C_ + cout) * C_ + cin] = (__bf16)w[idx];
}

// LayerNorm over channels + LeakyReLU(0.2), writes bf16 activation
__global__ __launch_bounds__(256)
void ln_lrelu_kernel(const float* __restrict__ in, const float* __restrict__ g,
                     const float* __restrict__ be, __bf16* __restrict__ outBF) {
    __shared__ float red[256];
    const int tid = threadIdx.x;
    const size_t base = (size_t)blockIdx.x * C_;
    const float v0 = in[base + tid];
    const float v1 = in[base + tid + 256];
    red[tid] = v0 + v1;
    __syncthreads();
    for (int s = 128; s > 0; s >>= 1) {
        if (tid < s) red[tid] += red[tid + s];
        __syncthreads();
    }
    const float mu = red[0] * (1.0f / C_);
    __syncthreads();
    const float d0 = v0 - mu, d1 = v1 - mu;
    red[tid] = d0 * d0 + d1 * d1;
    __syncthreads();
    for (int s = 128; s > 0; s >>= 1) {
        if (tid < s) red[tid] += red[tid + s];
        __syncthreads();
    }
    const float rstd = rsqrtf(red[0] * (1.0f / C_) + 1e-5f);
    float t0 = d0 * rstd * g[tid] + be[tid];
    float t1 = d1 * rstd * g[tid + 256] + be[tid + 256];
    t0 = t0 > 0.f ? t0 : 0.2f * t0;
    t1 = t1 > 0.f ? t1 : 0.2f * t1;
    outBF[base + tid]       = (__bf16)t0;
    outBF[base + tid + 256] = (__bf16)t1;
}

// depthwise causal conv (k=4) + SiLU over u half of uz
__global__ void dwconv_silu_kernel(const float* __restrict__ uz,
                                   const float* __restrict__ cw, const float* __restrict__ cb,
                                   float* __restrict__ uc, __bf16* __restrict__ ucbf) {
    const int idx = blockIdx.x * blockDim.x + threadIdx.x;  // B*L*DI
    const int d  = idx & (DI_ - 1);
    const int bl = idx >> 10;
    const int l  = bl & (L_ - 1);
    const int b  = bl >> 10;
    float acc = cb[d];
#pragma unroll
    for (int j = 0; j < DC_; ++j) {
        const int ll = l - (DC_ - 1) + j;
        if (ll >= 0)
            acc += cw[d * DC_ + j] * uz[((size_t)b * L_ + ll) * (2 * DI_) + d];
    }
    const float s = acc * (1.f / (1.f + __expf(-acc)));
    uc[idx]   = s;
    ucbf[idx] = (__bf16)s;
}

// selective-scan: h[n] = exp(delta*A[n])*h[n] + delta*B[n]*u ; y = sum h[n]*C[n]
// Shared B/C vectors double-buffered: one barrier per timestep.
__global__ __launch_bounds__(128)
void ssm_scan_kernel(const float* __restrict__ delta, const float* __restrict__ xdbc,
                     const float* __restrict__ uc, const float* __restrict__ A_log,
                     float* __restrict__ y) {
    __shared__ float sBC[2][32];
    const int tid = threadIdx.x;
    const int b = blockIdx.x >> 3;
    const int d = ((blockIdx.x & 7) << 7) + tid;
    float A[DS_], h[DS_];
#pragma unroll
    for (int n = 0; n < DS_; ++n) {
        A[n] = -__expf(A_log[(size_t)d * DS_ + n]);
        h[n] = 0.f;
    }
    if (tid < 32) sBC[0][tid] = xdbc[((size_t)b * L_) * 64 + 32 + tid];
    __syncthreads();
    for (int l = 0; l < L_; ++l) {
        const size_t bl = (size_t)b * L_ + l;
        if (l + 1 < L_ && tid < 32)
            sBC[(l + 1) & 1][tid] = xdbc[(bl + 1) * 64 + 32 + tid];
        const float* BC = sBC[l & 1];
        const float dl = delta[bl * DI_ + d];
        const float uv = uc[bl * DI_ + d];
        float acc = 0.f;
#pragma unroll
        for (int n = 0; n < DS_; ++n) {
            const float dA = __expf(dl * A[n]);
            h[n] = dA * h[n] + dl * BC[n] * uv;
            acc += h[n] * BC[16 + n];
        }
        y[bl * DI_ + d] = acc;
        __syncthreads();
    }
}

// y = (y + u*D) * silu(z), to bf16 for out_proj
__global__ void gate_kernel(const float* __restrict__ y, const float* __restrict__ uc,
                            const float* __restrict__ uz, const float* __restrict__ Dv,
                            __bf16* __restrict__ ybf) {
    const int idx = blockIdx.x * blockDim.x + threadIdx.x;  // B*L*DI
    const int d  = idx & (DI_ - 1);
    const int bl = idx >> 10;
    const float z = uz[(size_t)bl * (2 * DI_) + DI_ + d];
    const float g = z * (1.f / (1.f + __expf(-z)));
    ybf[idx] = (__bf16)((y[idx] + uc[idx] * Dv[d]) * g);
}

__global__ void reverse_bf_kernel(const __bf16* __restrict__ src, __bf16* __restrict__ dst) {
    const int idx = blockIdx.x * blockDim.x + threadIdx.x;  // B*L*C
    const int c  = idx & (C_ - 1);
    const int bl = idx >> 9;
    const int l  = bl & (L_ - 1);
    const int b  = bl >> 10;
    dst[idx] = src[((size_t)b * L_ + (L_ - 1 - l)) * C_ + c];
}

// place fo / reversed bo into comb [B][L][2C] bf16
__global__ void combine_kernel(const float* __restrict__ dirout, __bf16* __restrict__ comb, int rev) {
    const int idx = blockIdx.x * blockDim.x + threadIdx.x;  // B*L*C
    const int j  = idx & (C_ - 1);
    const int bl = idx >> 9;
    const int l  = bl & (L_ - 1);
    const int b  = bl >> 10;
    const int ls = rev ? (L_ - 1 - l) : l;
    comb[(size_t)bl * (2 * C_) + (rev ? C_ : 0) + j] =
        (__bf16)dirout[((size_t)b * L_ + ls) * C_ + j];
}

// ---------------------------------------------------------------------------
extern "C" void kernel_launch(void* const* d_in, const int* in_sizes, int n_in,
                              void* d_out, int out_size, void* d_ws, size_t ws_size,
                              hipStream_t stream) {
    (void)in_sizes; (void)n_in; (void)out_size; (void)ws_size;

    const int*   x      = (const int*)d_in[0];
    const float* emb    = (const float*)d_in[3];
    const float* conv_w = (const float*)d_in[4];
    const float* conv_b = (const float*)d_in[5];
    const float* ln_g   = (const float*)d_in[6];
    const float* ln_b   = (const float*)d_in[7];
    const float* proj_w = (const float*)d_in[8];
    const float* proj_b = (const float*)d_in[9];
    const float* p_in_w[2]   = {(const float*)d_in[10], (const float*)d_in[19]};
    const float* p_conv_w[2] = {(const float*)d_in[11], (const float*)d_in[20]};
    const float* p_conv_b[2] = {(const float*)d_in[12], (const float*)d_in[21]};
    const float* p_x_w[2]    = {(const float*)d_in[13], (const float*)d_in[22]};
    const float* p_dt_w[2]   = {(const float*)d_in[14], (const float*)d_in[23]};
    const float* p_dt_b[2]   = {(const float*)d_in[15], (const float*)d_in[24]};
    const float* p_A_log[2]  = {(const float*)d_in[16], (const float*)d_in[25]};
    const float* p_D[2]      = {(const float*)d_in[17], (const float*)d_in[26]};
    const float* p_out_w[2]  = {(const float*)d_in[18], (const float*)d_in[27]};

    // workspace carve-out (256B aligned slabs)
    char* ws = (char*)d_ws;
    size_t off = 0;
    auto carve = [&](size_t bytes) -> void* {
        void* p = ws + off;
        off = (off + bytes + 255) & ~(size_t)255;
        return p;
    };
    __bf16* wconvbf  = (__bf16*)carve((size_t)DEPTH_ * KT_ * C_ * C_ * 2);
    __bf16* inwbf    = (__bf16*)carve((size_t)2 * 2 * DI_ * C_ * 2);
    __bf16* xwbf     = (__bf16*)carve((size_t)2 * 64 * DI_ * 2);
    __bf16* dtwbf    = (__bf16*)carve((size_t)2 * DI_ * DTR_ * 2);
    __bf16* outwbf   = (__bf16*)carve((size_t)2 * C_ * DI_ * 2);
    __bf16* projwbf  = (__bf16*)carve((size_t)C_ * 2 * C_ * 2);
    __bf16* hbf      = (__bf16*)carve((size_t)NBL * C_ * 2);
    __bf16* hrevbf   = (__bf16*)carve((size_t)NBL * C_ * 2);
    float*  convtmp  = (float*)carve((size_t)NBL * C_ * 4);
    float*  uz       = (float*)carve((size_t)NBL * 2 * DI_ * 4);
    float*  ucf      = (float*)carve((size_t)NBL * DI_ * 4);
    __bf16* ucbf     = (__bf16*)carve((size_t)NBL * DI_ * 2);
    float*  xdbcf    = (float*)carve((size_t)NBL * 64 * 4);
    __bf16* xdbcbf   = (__bf16*)carve((size_t)NBL * 64 * 2);
    float*  deltaf   = (float*)carve((size_t)NBL * DI_ * 4);
    float*  yssm     = (float*)carve((size_t)NBL * DI_ * 4);
    __bf16* ybf      = (__bf16*)carve((size_t)NBL * DI_ * 2);
    float*  dirout   = (float*)carve((size_t)NBL * C_ * 4);
    __bf16* combbf   = (__bf16*)carve((size_t)NBL * 2 * C_ * 2);

    auto nb = [](size_t n) { return (unsigned)((n + 255) / 256); };

    // ---- weight conversions ----
    pack_conv_w_kernel<<<nb((size_t)DEPTH_ * C_ * C_ * KT_), 256, 0, stream>>>(conv_w, wconvbf);
    for (int dir = 0; dir < 2; ++dir) {
        f32_to_bf16_kernel<<<nb(2 * DI_ * C_), 256, 0, stream>>>(p_in_w[dir], inwbf + (size_t)dir * 2 * DI_ * C_, 2 * DI_ * C_);
        f32_to_bf16_kernel<<<nb(64 * DI_), 256, 0, stream>>>(p_x_w[dir], xwbf + (size_t)dir * 64 * DI_, 64 * DI_);
        f32_to_bf16_kernel<<<nb(DI_ * DTR_), 256, 0, stream>>>(p_dt_w[dir], dtwbf + (size_t)dir * DI_ * DTR_, DI_ * DTR_);
        f32_to_bf16_kernel<<<nb(C_ * DI_), 256, 0, stream>>>(p_out_w[dir], outwbf + (size_t)dir * C_ * DI_, C_ * DI_);
    }
    f32_to_bf16_kernel<<<nb(C_ * 2 * C_), 256, 0, stream>>>(proj_w, projwbf, C_ * 2 * C_);

    // ---- embedding ----
    embed_kernel<<<nb((size_t)NBL * C_), 256, 0, stream>>>(x, emb, hbf);

    // ---- conv stack: conv (async WMMA) -> LN+LeakyReLU -> bf16 activation ----
    for (int layer = 0; layer < DEPTH_; ++layer) {
        conv1d_wmma<<<dim3(C_ / BM_, L_ / BN_, B_), 256, GEMM_LDS_BYTES, stream>>>(
            wconvbf + (size_t)layer * KT_ * C_ * C_, hbf, conv_b + layer * C_, convtmp);
        ln_lrelu_kernel<<<NBL, 256, 0, stream>>>(convtmp, ln_g + layer * C_, ln_b + layer * C_, hbf);
    }

    // ---- bidirectional Mamba ----
    for (int dir = 0; dir < 2; ++dir) {
        const __bf16* hin = hbf;
        if (dir == 1) {
            reverse_bf_kernel<<<nb((size_t)NBL * C_), 256, 0, stream>>>(hbf, hrevbf);
            hin = hrevbf;
        }
        // in_proj: [2048 x 512] @ [512 x NBL]
        gemm_bt_wmma<<<dim3(2 * DI_ / BM_, NBL / BN_), 256, GEMM_LDS_BYTES, stream>>>(
            inwbf + (size_t)dir * 2 * DI_ * C_, C_, hin, C_, nullptr,
            uz, nullptr, 2 * DI_, NBL, C_, 0);
        // depthwise causal conv + SiLU
        dwconv_silu_kernel<<<nb((size_t)NBL * DI_), 256, 0, stream>>>(
            uz, p_conv_w[dir], p_conv_b[dir], ucf, ucbf);
        // x_proj: [64 x 1024] @ [1024 x NBL]
        gemm_bt_wmma<<<dim3(64 / BM_, NBL / BN_), 256, GEMM_LDS_BYTES, stream>>>(
            xwbf + (size_t)dir * 64 * DI_, DI_, ucbf, DI_, nullptr,
            xdbcf, xdbcbf, 64, NBL, DI_, 0);
        // delta = softplus(dt @ dt_w.T + dt_b): [1024 x 32] @ [32 x NBL]
        gemm_bt_wmma<<<dim3(DI_ / BM_, NBL / BN_), 256, GEMM_LDS_BYTES, stream>>>(
            dtwbf + (size_t)dir * DI_ * DTR_, DTR_, xdbcbf, 64, p_dt_b[dir],
            deltaf, nullptr, DI_, NBL, DTR_, 1);
        // selective scan
        ssm_scan_kernel<<<B_ * 8, 128, 0, stream>>>(deltaf, xdbcf, ucf, p_A_log[dir], yssm);
        // gating
        gate_kernel<<<nb((size_t)NBL * DI_), 256, 0, stream>>>(yssm, ucf, uz, p_D[dir], ybf);
        // out_proj: [512 x 1024] @ [1024 x NBL]
        gemm_bt_wmma<<<dim3(C_ / BM_, NBL / BN_), 256, GEMM_LDS_BYTES, stream>>>(
            outwbf + (size_t)dir * C_ * DI_, DI_, ybf, DI_, nullptr,
            dirout, nullptr, C_, NBL, DI_, 0);
        // pack into comb (reverse bo back to forward order)
        combine_kernel<<<nb((size_t)NBL * C_), 256, 0, stream>>>(dirout, combbf, dir);
    }

    // ---- final projection: [512 x 1024] @ [1024 x NBL] + bias -> d_out ----
    gemm_bt_wmma<<<dim3(C_ / BM_, NBL / BN_), 256, GEMM_LDS_BYTES, stream>>>(
        projwbf, 2 * C_, combbf, 2 * C_, proj_b,
        (float*)d_out, nullptr, C_, NBL, 2 * C_, 0);
}
